// LSTMActorCriticModel_16252156248566
// MI455X (gfx1250) — matrix operand order repaired and checked by reading, not compiled, not run on hardware
//
#include <hip/hip_runtime.h>

typedef __attribute__((ext_vector_type(16))) __bf16 v16bf;
typedef __attribute__((ext_vector_type(8)))  float  v8f;
typedef __attribute__((ext_vector_type(4)))  unsigned int uint4v;

union Frag { uint4v u[2]; v16bf v; };

__device__ __forceinline__ unsigned short f2bf(float f) {
  unsigned int u = __float_as_uint(f);
  unsigned int r = u + 0x7FFFu + ((u >> 16) & 1u);   // round-to-nearest-even
  return (unsigned short)(r >> 16);
}

// ---------------------------------------------------------------- fp32 -> bf16
__global__ void cvt_bf16(const float* __restrict__ in, unsigned short* __restrict__ out, int n) {
  int i = blockIdx.x * blockDim.x + threadIdx.x;
  if (i < n) out[i] = f2bf(in[i]);
}

// ---------------------------------------------------------------- WMMA GEMM
// Out[M,N] = act(A[M,K] @ W[N,K]^T + bias).  A,W bf16 row-major.
// Register-blocked: each wave computes a 32x64 output block
// (MT=2 x NT=4 tiles of 16x16) -> 8 WMMAs per 12 b128 loads per K-step.
template <bool RELU_BF16_OUT, int K>
__global__ __launch_bounds__(256) void gemm_wmma(
    const unsigned short* __restrict__ A,
    const unsigned short* __restrict__ W,
    const float* __restrict__ bias,
    void* __restrict__ Out,
    int M, int N)
{
  constexpr int MT = 2, NT = 4;                 // tiles per wave
  int wave = (int)((blockIdx.x * blockDim.x + threadIdx.x) >> 5);
  int lane = threadIdx.x & 31;
  int nbN  = N / (16 * NT);                     // N-blocks
  int bm = wave / nbN, bn = wave % nbN;
  if (bm >= M / (16 * MT)) return;

  int m16  = lane & 15;
  int half = lane >> 4;

  const unsigned short* Arow[MT];
#pragma unroll
  for (int i = 0; i < MT; i++)
    Arow[i] = A + (size_t)(bm * 32 + 16 * i + m16) * K;

  const unsigned short* Wrow[NT];
  int colN[NT];
#pragma unroll
  for (int j = 0; j < NT; j++) {
    colN[j] = bn * 64 + 16 * j + m16;
    Wrow[j] = W + (size_t)colN[j] * K;
  }

  v8f acc[MT][NT];
#pragma unroll
  for (int j = 0; j < NT; j++) {
    float bv = bias[colN[j]];
#pragma unroll
    for (int i = 0; i < MT; i++)
#pragma unroll
      for (int r = 0; r < 8; r++) acc[i][j][r] = bv;
  }

#pragma unroll 2
  for (int kk = 0; kk < K; kk += 32) {
    // stream-ahead prefetch (global_prefetch_b8); OOB speculative prefetch is dropped
    __builtin_prefetch(Arow[0] + kk + 512, 0, 1);
    __builtin_prefetch(Wrow[0] + kk + 512, 0, 1);

    Frag a[MT], b[NT];
#pragma unroll
    for (int i = 0; i < MT; i++) {
      // A 16x32 bf16 layout: lane m holds k = kk+8*half+[0..7] and kk+16+8*half+[0..7]
      a[i].u[0] = *(const uint4v*)(Arow[i] + kk + 8 * half);
      a[i].u[1] = *(const uint4v*)(Arow[i] + kk + 16 + 8 * half);
    }
#pragma unroll
    for (int j = 0; j < NT; j++) {
      // B = W^T, 32x16: lane (n=m16) holds k = kk+8*half+[0..7] and kk+16+8*half+[0..7] of W row n
      b[j].u[0] = *(const uint4v*)(Wrow[j] + kk + 8 * half);
      b[j].u[1] = *(const uint4v*)(Wrow[j] + kk + 16 + 8 * half);
    }
#pragma unroll
    for (int i = 0; i < MT; i++)
#pragma unroll
      for (int j = 0; j < NT; j++)
        acc[i][j] = __builtin_amdgcn_wmma_f32_16x16x32_bf16(
            false, a[i].v, false, b[j].v, (short)0, acc[i][j], false, false);
  }

#pragma unroll
  for (int i = 0; i < MT; i++)
#pragma unroll
    for (int j = 0; j < NT; j++)
#pragma unroll
      for (int r = 0; r < 8; r++) {
        int row = bm * 32 + 16 * i + r + 8 * half;
        size_t idx = (size_t)row * N + colN[j];
        float v = acc[i][j][r];
        if (RELU_BF16_OUT) {
          v = v > 0.f ? v : 0.f;
          ((unsigned short*)Out)[idx] = f2bf(v);
        } else {
          ((float*)Out)[idx] = v;
        }
      }
}

// ---------------------------------------------------------------- scan init
__global__ void scan_init(const float* __restrict__ hx, const float* __restrict__ cx,
                          float* __restrict__ hbuf, float* __restrict__ cbuf,
                          unsigned int* __restrict__ cnt) {
  int i = blockIdx.x * blockDim.x + threadIdx.x;
  if (i < 1024) { hbuf[i] = hx[i]; hbuf[1024 + i] = 0.f; cbuf[i] = cx[i]; }
  if (i == 0) *cnt = 0u;
}

// ---------------------------------------------------------------- LSTM scan
// 32 persistent blocks x 256 threads. Block b owns h[j], j in [32b, 32b+32).
// Its 128 W_hh rows (gates i,f,g,o) are pinned in LDS (256KB bf16, CDNA5 WGP).
#define GBLK 32
__global__ __launch_bounds__(256) void lstm_scan(
    const unsigned short* __restrict__ Whh,  // 4096 x 1024 bf16
    const float* __restrict__ xg,            // T x 4096 fp32 (includes b_ih)
    const float* __restrict__ b_hh,          // 4096
    float* __restrict__ hbuf,                // 2 x 1024 (double buffered)
    float* __restrict__ cbuf,                // 1024
    unsigned int* __restrict__ cnt,
    int T)
{
  extern __shared__ char smem[];
  unsigned short* Ws  = (unsigned short*)smem;                 // 128*1024 bf16
  float*          hsh = (float*)(smem + 128 * 1024 * 2);       // 1024
  float*          gsh = hsh + 1024;                            // 128 gate values

  int tid   = threadIdx.x;
  int jbase = blockIdx.x * 32;

  // Stage this block's W_hh slice: local row lr -> global row (lr>>5)*1024 + jbase + (lr&31)
  {
    unsigned int* Wsd = (unsigned int*)Ws;
    const unsigned int* Wgd = (const unsigned int*)Whh;
    for (int idx = tid; idx < 128 * 512; idx += 256) {
      int lr = idx >> 9, cd = idx & 511;
      int grow = (lr >> 5) * 1024 + jbase + (lr & 31);
      Wsd[(size_t)lr * 512 + cd] = Wgd[(size_t)grow * 512 + cd];
    }
  }

  int lr   = tid >> 1;            // local gate row (0..127)
  int hsel = tid & 1;             // which half of the 1024-dot
  int grow = (lr >> 5) * 1024 + jbase + (lr & 31);
  float bh = b_hh[grow];

  float c = 0.f;
  if (tid < 32) c = cbuf[jbase + tid];
  __syncthreads();

  for (int t = 0; t < T; t++) {
    const float* hsrc = hbuf + (t & 1) * 1024;
    for (int i = tid; i < 1024; i += 256) hsh[i] = hsrc[i];
    __syncthreads();

    const unsigned short* wr = Ws + (size_t)lr * 1024 + hsel * 512;
    const float* hh = hsh + hsel * 512;
    float acc = 0.f;
#pragma unroll 4
    for (int k = 0; k < 512; k += 8) {
      uint4v w4 = *(const uint4v*)(wr + k);
#pragma unroll
      for (int q = 0; q < 4; q++) {
        unsigned int wd = w4[q];
        acc += __uint_as_float((wd & 0xFFFFu) << 16) * hh[k + 2 * q];
        acc += __uint_as_float(wd & 0xFFFF0000u)     * hh[k + 2 * q + 1];
      }
    }
    float other = __shfl_xor(acc, 1, 32);
    if (hsel == 0)
      gsh[lr] = acc + other + xg[(size_t)t * 4096 + grow] + bh;
    __syncthreads();

    if (tid < 32) {
      float gi = gsh[tid], gf = gsh[32 + tid], gg = gsh[64 + tid], go = gsh[96 + tid];
      float si = 1.f / (1.f + __expf(-gi));
      float sf = 1.f / (1.f + __expf(-gf));
      float so = 1.f / (1.f + __expf(-go));
      c = sf * c + si * tanhf(gg);
      float h = so * tanhf(c);
      hbuf[((t + 1) & 1) * 1024 + jbase + tid] = h;
    }
    __syncthreads();
    __threadfence();
    if (tid == 0) {
      __hip_atomic_fetch_add(cnt, 1u, __ATOMIC_RELEASE, __HIP_MEMORY_SCOPE_AGENT);
      unsigned int target = (unsigned int)(t + 1) * GBLK;   // monotonic, reset by scan_init
      while (__hip_atomic_load(cnt, __ATOMIC_ACQUIRE, __HIP_MEMORY_SCOPE_AGENT) < target)
        __builtin_amdgcn_s_sleep(1);
    }
    __syncthreads();
    __threadfence();
  }
  if (tid < 32) cbuf[jbase + tid] = c;
}

// ---------------------------------------------------------------- head
__global__ __launch_bounds__(256) void head_kernel(
    const float* __restrict__ hbuf, const float* __restrict__ cbuf,
    const float* __restrict__ Wa, const float* __restrict__ ba,
    const float* __restrict__ Wc, const float* __restrict__ bc,
    float* __restrict__ out, int hsel)
{
  const float* h = hbuf + hsel * 1024;
  __shared__ float lg[19];
  int tid = threadIdx.x;
  if (tid < 19) {
    const float* w = (tid < 18) ? (Wa + (size_t)tid * 1024) : Wc;
    float s = (tid < 18) ? ba[tid] : bc[0];
    for (int k = 0; k < 1024; k++) s += w[k] * h[k];
    lg[tid] = s;
  }
  __syncthreads();
  if (tid == 0) {
    float m = lg[0];
    for (int a = 1; a < 18; a++) m = fmaxf(m, lg[a]);
    float ssum = 0.f;
    for (int a = 0; a < 18; a++) ssum += __expf(lg[a] - m);
    out[0] = lg[18];                                  // critic
    for (int a = 0; a < 18; a++) out[1 + a] = __expf(lg[a] - m) / ssum;
  }
  for (int i = tid; i < 1024; i += 256) {
    out[19 + i]        = h[i];
    out[19 + 1024 + i] = cbuf[i];
  }
}

// ---------------------------------------------------------------- launch
extern "C" void kernel_launch(void* const* d_in, const int* in_sizes, int n_in,
                              void* d_out, int out_size, void* d_ws, size_t ws_size,
                              hipStream_t stream) {
  (void)in_sizes; (void)n_in; (void)out_size; (void)ws_size;
  const int T = 8192, D = 512, H = 1024;

  const float* inputs   = (const float*)d_in[0];
  const float* hx       = (const float*)d_in[1];
  const float* cx       = (const float*)d_in[2];
  const float* W_fc1    = (const float*)d_in[3];
  const float* b_fc1    = (const float*)d_in[4];
  const float* W_ih     = (const float*)d_in[5];
  const float* b_ih     = (const float*)d_in[6];
  const float* W_hh     = (const float*)d_in[7];
  const float* b_hh     = (const float*)d_in[8];
  const float* W_actor  = (const float*)d_in[9];
  const float* b_actor  = (const float*)d_in[10];
  const float* W_critic = (const float*)d_in[11];
  const float* b_critic = (const float*)d_in[12];
  float* out = (float*)d_out;

  // workspace carve-up (bytes)
  char* ws = (char*)d_ws;
  size_t off = 0;
  auto alloc = [&](size_t bytes) { char* p = ws + off; off = (off + bytes + 255) & ~(size_t)255; return p; };
  unsigned short* inA_bf  = (unsigned short*)alloc((size_t)T * D * 2);        // 8 MB
  unsigned short* wfc1_bf = (unsigned short*)alloc((size_t)H * D * 2);        // 1 MB
  unsigned short* wih_bf  = (unsigned short*)alloc((size_t)4 * H * H * 2);    // 8 MB
  unsigned short* whh_bf  = (unsigned short*)alloc((size_t)4 * H * H * 2);    // 8 MB
  unsigned short* x_bf    = (unsigned short*)alloc((size_t)T * H * 2);        // 16 MB
  float*          xg      = (float*)alloc((size_t)T * 4 * H * 4);             // 128 MB
  float*          hbuf    = (float*)alloc(2 * H * 4);
  float*          cbuf    = (float*)alloc(H * 4);
  unsigned int*   cnt     = (unsigned int*)alloc(256);

  // 1) fp32 -> bf16 conversions
  cvt_bf16<<<(T * D + 255) / 256, 256, 0, stream>>>(inputs, inA_bf, T * D);
  cvt_bf16<<<(H * D + 255) / 256, 256, 0, stream>>>(W_fc1, wfc1_bf, H * D);
  cvt_bf16<<<(4 * H * H + 255) / 256, 256, 0, stream>>>(W_ih, wih_bf, 4 * H * H);
  cvt_bf16<<<(4 * H * H + 255) / 256, 256, 0, stream>>>(W_hh, whh_bf, 4 * H * H);

  // 2) x = relu(inputs @ W_fc1^T + b_fc1), bf16 out.
  //    32x64 per wave: (8192/32)*(1024/64) = 256*16 = 4096 waves -> 512 blocks
  gemm_wmma<true, 512><<<512, 256, 0, stream>>>(inA_bf, wfc1_bf, b_fc1, x_bf, T, H);

  // 3) xg = x @ W_ih^T + b_ih, fp32 out.
  //    (8192/32)*(4096/64) = 256*64 = 16384 waves -> 2048 blocks
  gemm_wmma<false, 1024><<<2048, 256, 0, stream>>>(x_bf, wih_bf, b_ih, xg, T, 4 * H);

  // 4) sequential LSTM scan (persistent, LDS-pinned W_hh, L2-atomic grid sync)
  scan_init<<<4, 256, 0, stream>>>(hx, cx, hbuf, cbuf, cnt);
  size_t smem = 128 * 1024 * 2 + 1024 * 4 + 128 * 4;   // 260.5 KB < 320 KB WGP LDS
  lstm_scan<<<GBLK, 256, smem, stream>>>(whh_bf, xg, b_hh, hbuf, cbuf, cnt, T);

  // 5) head: critic, softmax(actor), h, c  -> out[0], out[1..18], out[19..1042], out[1043..2066]
  head_kernel<<<1, 256, 0, stream>>>(hbuf, cbuf, W_actor, b_actor, W_critic, b_critic,
                                     out, T & 1);
}
